// SelfAttention_44367012168253
// MI455X (gfx1250) — compile-verified, matrix operand order checked by reading, not run on hardware
//
#include <hip/hip_runtime.h>
#include <cmath>

typedef _Float16 half_t;
typedef __attribute__((ext_vector_type(16))) _Float16 v16h;
typedef __attribute__((ext_vector_type(8)))  float    v8f;
typedef __attribute__((ext_vector_type(2)))  _Float16 h2;

__device__ __forceinline__ v8f wmma32(v16h a, v16h b, v8f c) {
  return __builtin_amdgcn_wmma_f32_16x16x32_f16(
      /*neg_a=*/false, a, /*neg_b=*/false, b,
      /*c_mod=*/(short)0, c, /*reuse_a=*/false, /*reuse_b=*/false);
}

// k index for half element pair m (m=0..7) of a 16-bit 16x32 A/B fragment.
// lane group g = lane>>4.  k(e) = g*8 + (e&7) + (e>=8 ? 16 : 0); pairs contiguous.
__device__ __forceinline__ int frag_k(int m, int g) {
  return ((m & 4) << 2) + g * 8 + ((m & 3) << 1);
}

// Load a 16x32 fragment whose per-lane data is contiguous along K at `base`
// (two 16-byte runs -> lowers to 2x ds_load_b128).
__device__ __forceinline__ v16h load_frag(const half_t* base, int g) {
  v16h f;
#pragma unroll
  for (int m = 0; m < 8; ++m) {
    h2 p = *(const h2*)(base + frag_k(m, g));
    f[2 * m] = p.x; f[2 * m + 1] = p.y;
  }
  return f;
}

// ---------------------------------------------------------------------------
// Tiled f16 WMMA GEMM:  C[M,N] = act(A[M,K] * W^T[N,K] + bias [+ add])
// A row-major f16 (lda=K), WT row-major [N,K] f16 (weights pre-transposed so
// both A and B fragments load as contiguous ds_load_b128 pairs).
// Double-buffered LDS pipeline: tile t+1 global->regs while tile t computes.
// Epilogue fully specialized at compile time (no pointer-null branches).
// OTRANS: f16 output written in [B,H,D,S] layout (attention V operand).
// ---------------------------------------------------------------------------
#define BM 128
#define BN 64
#define BK 32
#define LDA_ (BK + 8)   // 40 halves = 80B, multiple of 16B
#define LDBT (BK + 8)   // 40 halves

template <bool HAS_ADD, bool RELU, bool OUTF, bool OUTH, bool OTRANS>
__global__ __launch_bounds__(256) void gemm_wmma(
    const half_t* __restrict__ A, const half_t* __restrict__ WT,
    const float* __restrict__ bias, const float* add,
    float* outF, half_t* outH,
    int M, int N, int K)
{
  __shared__ half_t sA[2][BM * LDA_];
  __shared__ half_t sB[2][BN * LDBT];

  const int tid  = threadIdx.x;
  const int lane = tid & 31;
  const int wv   = tid >> 5;
  const int g    = lane >> 4;
  const int lr   = lane & 15;
  const int wm   = wv >> 1;     // 0..3
  const int wn   = wv & 1;      // 0..1

  const int nTilesN = N / BN;
  const int m0 = (blockIdx.x / nTilesN) * BM;
  const int n0 = (blockIdx.x % nTilesN) * BN;

  const int arow = tid >> 1;          // 0..127
  const int acol = (tid & 1) * 16;    // 0 or 16
  const int brow = tid >> 2;          // 0..63  (n within tile)
  const int bcol = (tid & 3) * 8;     // 0..24  (k within tile)

  const half_t* gA = A + (size_t)(m0 + arow) * K + acol;
  const half_t* gB = WT + (size_t)(n0 + brow) * K + bcol;

  // preload tile 0
  uint4 ra0 = ((const uint4*)gA)[0];
  uint4 ra1 = ((const uint4*)gA)[1];
  uint4 rb  = ((const uint4*)gB)[0];
  {
    uint4* la = (uint4*)(sA[0] + arow * LDA_ + acol);
    la[0] = ra0; la[1] = ra1;
    *(uint4*)(sB[0] + brow * LDBT + bcol) = rb;
  }
  __syncthreads();

  v8f c[2][2] = {};
  const int nT = K / BK;
  for (int t = 0; t < nT; ++t) {
    const int cur = t & 1;
    if (t + 1 < nT) {  // issue next tile's global loads before computing
      const half_t* pA = gA + (t + 1) * BK;
      const half_t* pB = gB + (t + 1) * BK;
      ra0 = ((const uint4*)pA)[0];
      ra1 = ((const uint4*)pA)[1];
      rb  = ((const uint4*)pB)[0];
      if (t + 2 < nT) {  // global_prefetch_b8 for t+2
        __builtin_prefetch(gA + (t + 2) * BK, 0, 1);
        __builtin_prefetch(gB + (t + 2) * BK, 0, 1);
      }
    }

    v16h a[2], b[2];
#pragma unroll
    for (int i = 0; i < 2; ++i)
      a[i] = load_frag(sA[cur] + (wm * 32 + i * 16 + lr) * LDA_, g);
#pragma unroll
    for (int j = 0; j < 2; ++j)
      b[j] = load_frag(sB[cur] + (wn * 32 + j * 16 + lr) * LDBT, g);
#pragma unroll
    for (int i = 0; i < 2; ++i)
#pragma unroll
      for (int j = 0; j < 2; ++j)
        c[i][j] = wmma32(a[i], b[j], c[i][j]);

    if (t + 1 < nT) {  // spill next tile into the alternate buffer
      uint4* la = (uint4*)(sA[cur ^ 1] + arow * LDA_ + acol);
      la[0] = ra0; la[1] = ra1;
      *(uint4*)(sB[cur ^ 1] + brow * LDBT + bcol) = rb;
    }
    __syncthreads();
  }

  // branch-free specialized epilogue: rows within a fragment are consecutive,
  // so all output indices are a base + linear walk.
#pragma unroll
  for (int i = 0; i < 2; ++i) {
#pragma unroll
    for (int j = 0; j < 2; ++j) {
      const int n    = n0 + wn * 32 + j * 16 + lr;
      const int mrow = m0 + wm * 32 + i * 16 + 8 * g;
      const float bv = bias[n];
      size_t idx = (size_t)mrow * N + n;
      size_t tix = 0;
      if constexpr (OTRANS) {
        // mrow = b*512 + s ; n = h*64 + d -> [B,H,D,S]; walk is +1 in s
        const int bb = mrow >> 9, ss = mrow & 511, hh = n >> 6, dd = n & 63;
        tix = (((size_t)bb * 8 + hh) * 64 + dd) * 512 + ss;
      }
#pragma unroll
      for (int r = 0; r < 8; ++r) {
        float v = c[i][j][r] + bv;
        if constexpr (HAS_ADD) v += add[idx];
        if constexpr (RELU)    v = fmaxf(v, 0.0f);
        if constexpr (OUTF)    outF[idx] = v;
        if constexpr (OUTH) {
          if constexpr (OTRANS) { outH[tix] = (half_t)v; tix += 1; }
          else                  outH[idx] = (half_t)v;
        }
        idx += N;
      }
    }
  }
}

// ---------------------------------------------------------------------------
// Fused flash attention: q,k in [B,S,H*D] f16, v in [B,H,D,S] f16 (transposed
// by the V GEMM epilogue so V fragments load contiguously), o in [B,S,H*D].
// Block = 4 waves = 64 query rows of one (b,h). Online softmax, WMMA for
// both QK^T and P*V.  blockIdx.x = (b*H + h)*(S/64) + qb.
// ---------------------------------------------------------------------------
#define HD_ 512
#define LKV 72   // K tile row stride (32 keys x 64 d)
#define LVT 40   // V tile row stride (64 d x 32 keys)
#define LP  40   // P staging row stride

__global__ __launch_bounds__(128) void attn_kernel(
    const half_t* __restrict__ q, const half_t* __restrict__ k,
    const half_t* __restrict__ v, half_t* __restrict__ o, float scale)
{
  __shared__ half_t sK[32 * LKV];
  __shared__ half_t sV[64 * LVT];
  __shared__ half_t sP[4 * 16 * LP];

  const int tid  = threadIdx.x;
  const int lane = tid & 31;
  const int wv   = tid >> 5;
  const int g    = lane >> 4;
  const int lr   = lane & 15;

  const int S_ = 512;
  const int qb = blockIdx.x & 7;           // S/64 = 8
  const int bh = blockIdx.x >> 3;          // b*8 + h
  const int h  = bh & 7;
  const int b  = bh >> 3;
  const size_t rowBase = ((size_t)b * S_) * HD_ + (size_t)h * 64;
  const size_t vBase   = ((size_t)bh * 64) * S_;   // [B,H,D,S]

  // Q fragments: 16 rows x 64 d -> two 16x32 A fragments
  const int qrow = qb * 64 + wv * 16 + lr;
  v16h qa[2];
#pragma unroll
  for (int kk = 0; kk < 2; ++kk)
    qa[kk] = load_frag(q + rowBase + (size_t)qrow * HD_ + kk * 32, g);

  v8f   oacc[4] = {};
  float rmax[8], rsum[8];
#pragma unroll
  for (int r = 0; r < 8; ++r) { rmax[r] = -1e30f; rsum[r] = 0.0f; }

  half_t* myP = sP + wv * 16 * LP;

  for (int kt = 0; kt < S_ / 32; ++kt) {
    // K tile: 32 keys x 64 d (rows contiguous in d)
    {
      const int r0 = tid >> 3;            // 0..15
      const int c0 = (tid & 7) * 8;
#pragma unroll
      for (int rr = 0; rr < 2; ++rr) {
        const int krow = r0 + rr * 16;
        *(uint4*)(sK + krow * LKV + c0) = *(const uint4*)(
            k + rowBase + (size_t)(kt * 32 + krow) * HD_ + c0);
      }
      // V tile: 64 d-rows x 32 keys (rows contiguous in s)
      const int dv = tid >> 1;            // 0..63
      const int cv = (tid & 1) * 16;      // 0 or 16
      const uint4* gv = (const uint4*)(v + vBase + (size_t)dv * S_ + kt * 32 + cv);
      uint4* lv = (uint4*)(sV + dv * LVT + cv);
      lv[0] = gv[0];
      lv[1] = gv[1];
    }
    __syncthreads();

    // scores: 16 queries x 32 keys, K-dim = 64 (two chunks of 32)
    v8f sf[2] = {};
#pragma unroll
    for (int nt = 0; nt < 2; ++nt) {
#pragma unroll
      for (int kk = 0; kk < 2; ++kk) {
        v16h bf = load_frag(sK + (nt * 16 + lr) * LKV + kk * 32, g);
        sf[nt] = wmma32(qa[kk], bf, sf[nt]);
      }
    }

    // online softmax per row (row r+8g; values striped across 16 lanes)
#pragma unroll
    for (int r = 0; r < 8; ++r) {
      const float s0 = sf[0][r] * scale;
      const float s1 = sf[1][r] * scale;
      float tm = fmaxf(s0, s1);
#pragma unroll
      for (int msk = 8; msk >= 1; msk >>= 1)
        tm = fmaxf(tm, __shfl_xor(tm, msk, 32));
      const float nm    = fmaxf(rmax[r], tm);
      const float alpha = __expf(rmax[r] - nm);
      const float p0 = __expf(s0 - nm);
      const float p1 = __expf(s1 - nm);
      float ps = p0 + p1;
#pragma unroll
      for (int msk = 8; msk >= 1; msk >>= 1)
        ps += __shfl_xor(ps, msk, 32);
      rsum[r] = rsum[r] * alpha + ps;
      rmax[r] = nm;
#pragma unroll
      for (int j = 0; j < 4; ++j) oacc[j][r] *= alpha;
      myP[(r + 8 * g) * LP + lr]      = (half_t)p0;
      myP[(r + 8 * g) * LP + 16 + lr] = (half_t)p1;
    }

    // reload P as a 16x32 A fragment (LDS ops stay in order within a wave)
    v16h pf = load_frag(myP + lr * LP, g);

    // O += P (16x32) x V (32x64): four 16x16 n-subtiles
#pragma unroll
    for (int j = 0; j < 4; ++j) {
      v16h vf = load_frag(sV + (j * 16 + lr) * LVT, g);
      oacc[j] = wmma32(pf, vf, oacc[j]);
    }
    __syncthreads();
  }

#pragma unroll
  for (int j = 0; j < 4; ++j) {
    const int d = j * 16 + lr;
    half_t* po = o + rowBase + (size_t)(qb * 64 + wv * 16 + 8 * g) * HD_ + d;
#pragma unroll
    for (int r = 0; r < 8; ++r) {
      *po = (half_t)(oacc[j][r] / rsum[r]);
      po += HD_;
    }
  }
}

// ---------------------------------------------------------------------------
// Elementwise / reduction kernels
// ---------------------------------------------------------------------------
__global__ __launch_bounds__(256) void embed_kernel(
    const int* __restrict__ tokens, const float* __restrict__ emb,
    float* __restrict__ x, half_t* __restrict__ xh)
{
  const int row = blockIdx.x, tid = threadIdx.x;
  const int tok = tokens[row];
  const float mask = (tok != 0) ? 1.0f : 0.0f;
  const float* er = emb + (size_t)tok * 512;
  const float v0 = er[tid] * mask;
  const float v1 = er[tid + 256] * mask;
  const size_t o = (size_t)row * 512;
  x[o + tid] = v0;        x[o + tid + 256] = v1;
  xh[o + tid] = (half_t)v0; xh[o + tid + 256] = (half_t)v1;
}

// residual add + l2 normalize; writes f32 (in place over x) and f16 copy
__global__ __launch_bounds__(256) void l2norm_res(
    const float* x, const float* __restrict__ y, float* xo, half_t* xh)
{
  __shared__ float red[256];
  const int row = blockIdx.x, tid = threadIdx.x;
  const size_t o = (size_t)row * 512;
  const float v0 = x[o + tid] + y[o + tid];
  const float v1 = x[o + tid + 256] + y[o + tid + 256];
  red[tid] = v0 * v0 + v1 * v1;
  __syncthreads();
  for (int s = 128; s > 0; s >>= 1) {
    if (tid < s) red[tid] += red[tid + s];
    __syncthreads();
  }
  const float nrm = rsqrtf(fmaxf(red[0], 1e-12f));
  xo[o + tid] = v0 * nrm;        xo[o + tid + 256] = v1 * nrm;
  xh[o + tid] = (half_t)(v0 * nrm); xh[o + tid + 256] = (half_t)(v1 * nrm);
}

// Conv1D(kernel=5, stride=5, 1 out channel): blockIdx.x = b*102 + t
__global__ __launch_bounds__(256) void conv_pool(
    const float* __restrict__ x, const float* __restrict__ Wc,
    const float* __restrict__ bc, float* __restrict__ pooled)
{
  __shared__ float red[256];
  const int bi = blockIdx.x / 102, t = blockIdx.x % 102;
  const int tid = threadIdx.x;
  float acc = 0.0f;
  for (int i = tid; i < 5 * 512; i += 256) {
    const int wi = i >> 9, e = i & 511;
    acc += x[((size_t)bi * 512 + (t * 5 + wi)) * 512 + e] * Wc[wi * 512 + e];
  }
  red[tid] = acc;
  __syncthreads();
  for (int s = 128; s > 0; s >>= 1) {
    if (tid < s) red[tid] += red[tid + s];
    __syncthreads();
  }
  if (tid == 0) pooled[bi * 102 + t] = red[0] + bc[0];
}

__global__ __launch_bounds__(256) void dense_softmax(
    const float* __restrict__ pooled, const float* __restrict__ Wd,
    const float* __restrict__ bd, float* __restrict__ out)
{
  __shared__ float sp[102];
  __shared__ float logit[1000];
  __shared__ float red[256];
  const int b = blockIdx.x, tid = threadIdx.x;
  if (tid < 102) sp[tid] = pooled[b * 102 + tid];
  __syncthreads();
  for (int oi = tid; oi < 1000; oi += 256) {
    float acc = bd[oi];
    for (int t = 0; t < 102; ++t) acc += sp[t] * Wd[t * 1000 + oi];
    logit[oi] = acc;
  }
  __syncthreads();
  float m = -1e30f;
  for (int oi = tid; oi < 1000; oi += 256) m = fmaxf(m, logit[oi]);
  red[tid] = m;
  __syncthreads();
  for (int s = 128; s > 0; s >>= 1) {
    if (tid < s) red[tid] = fmaxf(red[tid], red[tid + s]);
    __syncthreads();
  }
  m = red[0];
  __syncthreads();
  float sum = 0.0f;
  for (int oi = tid; oi < 1000; oi += 256) {
    const float e = __expf(logit[oi] - m);
    logit[oi] = e;
    sum += e;
  }
  red[tid] = sum;
  __syncthreads();
  for (int s = 128; s > 0; s >>= 1) {
    if (tid < s) red[tid] += red[tid + s];
    __syncthreads();
  }
  const float inv = 1.0f / red[0];
  for (int oi = tid; oi < 1000; oi += 256)
    out[(size_t)b * 1000 + oi] = logit[oi] * inv;
}

// weight conversions — all weights stored TRANSPOSED [N,K] so GEMM B-tiles
// and fragments load contiguously.
__global__ __launch_bounds__(256) void convert_qkv_t(
    const float* __restrict__ in, half_t* __restrict__ outw, int total)
{
  const int i = blockIdx.x * 256 + threadIdx.x;
  if (i >= total) return;
  // in: [L,H,E,D] -> out: [L, N=H*D, K=E]
  const int d = i & 63;
  const int e = (i >> 6) & 511;
  const int hh = (i >> 15) & 7;
  const int l = i >> 18;
  outw[((size_t)l * 512 + hh * 64 + d) * 512 + e] = (half_t)in[i];
}

__global__ __launch_bounds__(256) void convert_trans(
    const float* __restrict__ in, half_t* __restrict__ outw,
    int K, int N, int total)
{
  const int i = blockIdx.x * 256 + threadIdx.x;
  if (i >= total) return;
  // in: [L,K,N] -> out: [L,N,K]
  const int n = i % N;
  const int kk = (i / N) % K;
  const int l = i / (N * K);
  outw[((size_t)l * N + n) * K + kk] = (half_t)in[i];
}

// ---------------------------------------------------------------------------
extern "C" void kernel_launch(void* const* d_in, const int* in_sizes, int n_in,
                              void* d_out, int out_size, void* d_ws, size_t ws_size,
                              hipStream_t stream) {
  (void)in_sizes; (void)n_in; (void)out_size; (void)ws_size;
  const int B_ = 64, S_ = 512, E_ = 512, H_ = 8, D_ = 64, L_ = 4, LOUT = 102;
  const int M = B_ * S_;  // 32768

  const int*   tokens = (const int*)d_in[0];
  const float* emb = (const float*)d_in[1];
  const float* Wq = (const float*)d_in[2];  const float* bq = (const float*)d_in[3];
  const float* Wk = (const float*)d_in[4];  const float* bk = (const float*)d_in[5];
  const float* Wv = (const float*)d_in[6];  const float* bv = (const float*)d_in[7];
  const float* Wo = (const float*)d_in[8];  const float* bo = (const float*)d_in[9];
  const float* W1 = (const float*)d_in[10]; const float* b1 = (const float*)d_in[11];
  const float* W2 = (const float*)d_in[12]; const float* b2 = (const float*)d_in[13];
  const float* Wc = (const float*)d_in[14]; const float* bc = (const float*)d_in[15];
  const float* Wd = (const float*)d_in[16]; const float* bd = (const float*)d_in[17];
  float* out = (float*)d_out;

  char* ws = (char*)d_ws;
  size_t off = 0;
  auto alloc = [&](size_t bytes) -> char* {
    char* p = ws + off;
    off += (bytes + 255) & ~(size_t)255;
    return p;
  };
  float*  x   = (float*)alloc((size_t)M * E_ * 4);        // 67 MB
  half_t* xh  = (half_t*)alloc((size_t)M * E_ * 2);       // 33.5 MB
  half_t* qh  = (half_t*)alloc((size_t)M * E_ * 2);
  half_t* kh  = (half_t*)alloc((size_t)M * E_ * 2);
  half_t* vt  = (half_t*)alloc((size_t)M * E_ * 2);       // [B,H,D,S]
  half_t* ah  = (half_t*)alloc((size_t)M * E_ * 2);
  float*  y   = (float*)alloc((size_t)M * E_ * 4);
  half_t* f1  = (half_t*)alloc((size_t)M * 2 * E_ * 2);   // 67 MB
  half_t* wqh = (half_t*)alloc((size_t)L_ * E_ * 512 * 2);
  half_t* wkh = (half_t*)alloc((size_t)L_ * E_ * 512 * 2);
  half_t* wvh = (half_t*)alloc((size_t)L_ * E_ * 512 * 2);
  half_t* woh = (half_t*)alloc((size_t)L_ * 512 * 512 * 2);
  half_t* w1h = (half_t*)alloc((size_t)L_ * 512 * 1024 * 2);
  half_t* w2h = (half_t*)alloc((size_t)L_ * 1024 * 512 * 2);
  float*  pooled = (float*)alloc((size_t)B_ * LOUT * 4);

  const int qkvTot = L_ * H_ * E_ * D_;  // 1,048,576
  convert_qkv_t<<<(qkvTot + 255) / 256, 256, 0, stream>>>(Wq, wqh, qkvTot);
  convert_qkv_t<<<(qkvTot + 255) / 256, 256, 0, stream>>>(Wk, wkh, qkvTot);
  convert_qkv_t<<<(qkvTot + 255) / 256, 256, 0, stream>>>(Wv, wvh, qkvTot);
  const int woTot = L_ * 512 * 512;
  convert_trans<<<(woTot + 255) / 256, 256, 0, stream>>>(Wo, woh, 512, 512, woTot);
  const int w1Tot = L_ * 512 * 1024;
  convert_trans<<<(w1Tot + 255) / 256, 256, 0, stream>>>(W1, w1h, 512, 1024, w1Tot);
  convert_trans<<<(w1Tot + 255) / 256, 256, 0, stream>>>(W2, w2h, 1024, 512, w1Tot);

  embed_kernel<<<M, 256, 0, stream>>>(tokens, emb, x, xh);

  const float scale = 1.0f / sqrtf(512.0f);  // reference scales by 1/sqrt(E)
  const int g512  = (M / BM) * (512 / BN);
  const int g1024 = (M / BM) * (1024 / BN);

  for (int i = 0; i < L_; ++i) {
    // QKV projections -> f16 (V written transposed [B,H,D,S])
    gemm_wmma<false, false, false, true, false><<<g512, 256, 0, stream>>>(
        xh, wqh + (size_t)i * 512 * 512, bq + i * 512, nullptr, nullptr, qh,
        M, 512, 512);
    gemm_wmma<false, false, false, true, false><<<g512, 256, 0, stream>>>(
        xh, wkh + (size_t)i * 512 * 512, bk + i * 512, nullptr, nullptr, kh,
        M, 512, 512);
    gemm_wmma<false, false, false, true, true><<<g512, 256, 0, stream>>>(
        xh, wvh + (size_t)i * 512 * 512, bv + i * 512, nullptr, nullptr, vt,
        M, 512, 512);
    attn_kernel<<<B_ * H_ * (S_ / 64), 128, 0, stream>>>(qh, kh, vt, ah, scale);
    // output projection -> f32
    gemm_wmma<false, false, true, false, false><<<g512, 256, 0, stream>>>(
        ah, woh + (size_t)i * 512 * 512, bo + i * 512, nullptr, y, nullptr,
        M, 512, 512);
    l2norm_res<<<M, 256, 0, stream>>>(x, y, x, xh);
    // FFN: relu GEMM -> f16, then residual GEMM -> f32 + f16
    gemm_wmma<false, true, false, true, false><<<g1024, 256, 0, stream>>>(
        xh, w1h + (size_t)i * 512 * 1024, b1 + i * 1024, nullptr, nullptr, f1,
        M, 1024, 512);
    gemm_wmma<true, false, true, true, false><<<g512, 256, 0, stream>>>(
        f1, w2h + (size_t)i * 1024 * 512, b2 + i * 512, x, x, xh,
        M, 512, 1024);
  }

  conv_pool<<<B_ * LOUT, 256, 0, stream>>>(x, Wc, bc, pooled);
  dense_softmax<<<B_, 256, 0, stream>>>(pooled, Wd, bd, out);
}
